// SimpleSubModule_12446815223957
// MI455X (gfx1250) — 8-bit matrix operand order repaired and checked by reading, compiled, not run on hardware
//
#include <hip/hip_runtime.h>
#include <stdint.h>

#define M_ROWS  8192
#define K_DIM   4096
#define N_COLS  4096

#define BM 256
#define BN 128
#define BK 64
#define LDS_STRIDE 80
#define NKT (K_DIM / BK)

typedef __attribute__((ext_vector_type(8))) int v8i;

__device__ __forceinline__ void async_copy16(unsigned lds_off,
                                             const signed char* g) {
    asm volatile("global_load_async_to_lds_b128 %0, %1, off"
                 :: "v"(lds_off), "v"(g)
                 : "memory");
}
__device__ __forceinline__ void wait_async0() {
    asm volatile("s_wait_asynccnt 0x0" ::: "memory");
}

__global__ void init_stats(unsigned int* stats) {
    if (threadIdx.x < 2) stats[threadIdx.x] = 0u;
}

__global__ void absmax_kernel(const float* __restrict__ v, size_t n4,
                              unsigned int* __restrict__ stat) {
    __shared__ unsigned int smax;
    if (threadIdx.x == 0) smax = 0u;
    __syncthreads();

    float m = 0.0f;
    size_t idx    = (size_t)blockIdx.x * blockDim.x + threadIdx.x;
    size_t stride = (size_t)gridDim.x * blockDim.x;
    const float4* v4 = (const float4*)v;
    for (size_t i = idx; i < n4; i += stride) {
        float4 a = v4[i];
        m = fmaxf(m, fabsf(a.x));
        m = fmaxf(m, fabsf(a.y));
        m = fmaxf(m, fabsf(a.z));
        m = fmaxf(m, fabsf(a.w));
    }
    atomicMax(&smax, __float_as_uint(m));
    __syncthreads();
    if (threadIdx.x == 0) atomicMax(stat, smax);
}

__global__ void quant_kernel(const float* __restrict__ v,
                             unsigned int* __restrict__ q4, size_t n4,
                             const unsigned int* __restrict__ stat,
                             float qmin, float qmax) {
    float am = __uint_as_float(*stat);
    float s  = (am > 0.0f) ? (am / 127.0f) : 1.0f;

    size_t idx    = (size_t)blockIdx.x * blockDim.x + threadIdx.x;
    size_t stride = (size_t)gridDim.x * blockDim.x;
    const float4* v4 = (const float4*)v;
    for (size_t i = idx; i < n4; i += stride) {
        float4 a = v4[i];
        int q0 = (int)fminf(fmaxf(rintf(a.x / s), qmin), qmax);
        int q1 = (int)fminf(fmaxf(rintf(a.y / s), qmin), qmax);
        int q2 = (int)fminf(fmaxf(rintf(a.z / s), qmin), qmax);
        int q3 = (int)fminf(fmaxf(rintf(a.w / s), qmin), qmax);
        q4[i] = (unsigned int)(q0 & 255) | ((unsigned int)(q1 & 255) << 8) |
                ((unsigned int)(q2 & 255) << 16) | ((unsigned int)(q3 & 255) << 24);
    }
}

__launch_bounds__(256)
__global__ void gemm_i8_wmma(const signed char* __restrict__ qx,
                             const signed char* __restrict__ qw,
                             const float* __restrict__ bias,
                             const unsigned int* __restrict__ stats,
                             float* __restrict__ out) {
    __shared__ __align__(16) unsigned char As[2][BM * LDS_STRIDE];
    __shared__ __align__(16) unsigned char Bs[2][BN * LDS_STRIDE];

    const int tid    = threadIdx.x;
    const int lane   = tid & 31;
    const int wave   = tid >> 5;
    const int wmBase = (wave >> 1) * 64;
    const int wnBase = (wave & 1) * 64;
    const int m0 = blockIdx.y * BM;
    const int n0 = blockIdx.x * BN;

    const int laneRow = lane & 15;
    const int kbA = (lane >> 4) * 16;
    const int kbB = (lane >> 4) * 16;

    const unsigned ldsA[2] = { (unsigned)(uintptr_t)&As[0][0],
                               (unsigned)(uintptr_t)&As[1][0] };
    const unsigned ldsB[2] = { (unsigned)(uintptr_t)&Bs[0][0],
                               (unsigned)(uintptr_t)&Bs[1][0] };

    v8i acc[4][4];
#pragma unroll
    for (int i = 0; i < 4; ++i)
#pragma unroll
        for (int j = 0; j < 4; ++j)
            acc[i][j] = (v8i){0, 0, 0, 0, 0, 0, 0, 0};

    auto stage = [&](int kk, int b) {
#pragma unroll
        for (int it = 0; it < 4; ++it) {
            int chunk = tid + it * 256;
            int r = chunk >> 2;
            int c = (chunk & 3) * 16;
            async_copy16(ldsA[b] + (unsigned)(r * LDS_STRIDE + c),
                         qx + (size_t)(m0 + r) * K_DIM + kk + c);
        }
#pragma unroll
        for (int it = 0; it < 2; ++it) {
            int chunk = tid + it * 256;
            int r = chunk >> 2;
            int c = (chunk & 3) * 16;
            async_copy16(ldsB[b] + (unsigned)(r * LDS_STRIDE + c),
                         qw + (size_t)(n0 + r) * K_DIM + kk + c);
        }
    };

    stage(0, 0);

    for (int kt = 0; kt < NKT; ++kt) {
        const int cur = kt & 1;
        wait_async0();
        __syncthreads();
        if (kt + 1 < NKT)
            stage((kt + 1) * BK, cur ^ 1);

        const unsigned char* Ab = &As[cur][0];
        const unsigned char* Bb = &Bs[cur][0];

        v8i afrag[4];
#pragma unroll
        for (int tm = 0; tm < 4; ++tm) {
            const unsigned char* rp =
                Ab + (wmBase + tm * 16 + laneRow) * LDS_STRIDE;
            union { v8i v; unsigned long long q[4]; } f;
            f.q[0] = *(const unsigned long long*)(rp + kbA);
            f.q[1] = *(const unsigned long long*)(rp + kbA + 8);
            f.q[2] = *(const unsigned long long*)(rp + kbA + 32);
            f.q[3] = *(const unsigned long long*)(rp + kbA + 40);
            afrag[tm] = f.v;
        }
#pragma unroll
        for (int tn = 0; tn < 4; ++tn) {
            const unsigned char* rp =
                Bb + (wnBase + tn * 16 + laneRow) * LDS_STRIDE;
            union { v8i v; uint4 u[2]; } f;
            f.u[0] = *(const uint4*)(rp + kbB);
            f.u[1] = *(const uint4*)(rp + kbB + 32);
            v8i bfrag = f.v;
#pragma unroll
            for (int tm = 0; tm < 4; ++tm)
                acc[tm][tn] = __builtin_amdgcn_wmma_i32_16x16x64_iu8(
                     true, afrag[tm],  true, bfrag,
                    acc[tm][tn],  false,  false);
        }
    }

    float amx = __uint_as_float(stats[0]);
    float amw = __uint_as_float(stats[1]);
    float sx = (amx > 0.0f ? amx : 127.0f) / 127.0f;
    float sw = (amw > 0.0f ? amw : 127.0f) / 127.0f;
    float scale = 4.0f * sx * sw;

    const int mrowBase = (lane >> 4) * 8;
#pragma unroll
    for (int tn = 0; tn < 4; ++tn) {
        int col = n0 + wnBase + tn * 16 + laneRow;
        float b4 = 4.0f * bias[col];
#pragma unroll
        for (int tm = 0; tm < 4; ++tm) {
            int rowBase = m0 + wmBase + tm * 16 + mrowBase;
#pragma unroll
            for (int j = 0; j < 8; ++j) {
                out[(size_t)(rowBase + j) * N_COLS + col] =
                    scale * (float)acc[tm][tn][j] + b4;
            }
        }
    }
}

extern "C" void kernel_launch(void* const* d_in, const int* in_sizes, int n_in,
                              void* d_out, int out_size, void* d_ws, size_t ws_size,
                              hipStream_t stream) {
    const float* x    = (const float*)d_in[0];
    const float* W    = (const float*)d_in[1];
    const float* bias = (const float*)d_in[2];
    float* out = (float*)d_out;

    unsigned int* stats = (unsigned int*)d_ws;
    signed char* qx = (signed char*)d_ws + 256;
    signed char* qw = qx + (size_t)M_ROWS * K_DIM;

    size_t nx4 = (size_t)M_ROWS * K_DIM / 4;
    size_t nw4 = (size_t)N_COLS * K_DIM / 4;

    init_stats<<<1, 32, 0, stream>>>(stats);
    absmax_kernel<<<2048, 256, 0, stream>>>(x, nx4, &stats[0]);
    absmax_kernel<<<1024, 256, 0, stream>>>(W, nw4, &stats[1]);
    quant_kernel<<<2048, 256, 0, stream>>>(x, (unsigned int*)qx, nx4, &stats[0],
                                           -128.0f, 127.0f);
    quant_kernel<<<1024, 256, 0, stream>>>(W, (unsigned int*)qw, nw4, &stats[1],
                                           -127.0f, 127.0f);

    dim3 grid(N_COLS / BN, M_ROWS / BM);
    gemm_i8_wmma<<<grid, 256, 0, stream>>>(qx, qw, bias, stats, out);
}
